// VGCN_2_28346784154175
// MI455X (gfx1250) — matrix operand order repaired and checked by reading, not compiled, not run on hardware
//
#include <hip/hip_runtime.h>

// ---------------------------------------------------------------------------
// VGCN forward for MI455X (gfx1250, wave32).
// Memory-bound workload (~120MB traffic, <2.5 GFLOP) -> fuse heads, one pass
// per tensor, WMMA f16->f32 for the dense GEMMs, L2-resident atomics for SpMM.
// ---------------------------------------------------------------------------

typedef __attribute__((ext_vector_type(16))) _Float16 v16h;
typedef __attribute__((ext_vector_type(8)))  float    v8f;

#define NNODE 50000
#define NFEAT 256
#define NHID  64
#define NCLS  40
#define NHEAD 80              // [W11 | W12] concatenated columns
#define NEDGE 800000
#define MTILE ((NNODE + 15) / 16)   // 3125 row tiles of 16

// ---------------------------------------------------------------------------
// Zero the two scatter accumulators (contiguous region of n floats).
// ---------------------------------------------------------------------------
__global__ __launch_bounds__(256) void k_zero(float* __restrict__ p, long n) {
  long i = ((long)blockIdx.x * blockDim.x + threadIdx.x) * 4;
  if (i + 3 < n) {
    *(float4*)(p + i) = make_float4(0.f, 0.f, 0.f, 0.f);
  } else {
    for (long j = i; j < n; ++j) p[j] = 0.f;
  }
}

// ---------------------------------------------------------------------------
// GEMM1: xw[N,64] = x[N,256] @ W1[256,64]  via v_wmma_f32_16x16x32_f16.
// One wave computes a 16x64 tile (4 n-tiles). W1 staged in LDS column-major
// as f16 so a B fragment is one 32B-aligned contiguous LDS read.
// A fragment as built here (contiguous k order, same as B): lane<16 holds row M=lane, K = k0+{0..15};
// lane>=16 holds K = k0+16+{0..15}.
// ---------------------------------------------------------------------------
__global__ __launch_bounds__(256) void k_gemm1(const float* __restrict__ x,
                                               const float* __restrict__ W1,
                                               float* __restrict__ xw) {
  __shared__ __align__(32) _Float16 Wt[NHID * NFEAT];  // Wt[c*256 + r]
  const int tid = threadIdx.x;
  for (int i = tid; i < NHID * NFEAT; i += 256) {      // coalesced W1 read
    int r = i >> 6, c = i & 63;
    Wt[c * NFEAT + r] = (_Float16)W1[i];
  }
  __syncthreads();

  const int wave = tid >> 5, lane = tid & 31;
  const int tile = blockIdx.x * 8 + wave;
  if (tile >= MTILE) return;                           // uniform per wave

  const int lm  = lane & 15;
  const int hi  = lane >> 4;          // 0: lanes 0-15, 1: lanes 16-31
  const int klo = hi * 8;
  const int row = tile * 16 + lm;
  const float* xrow = x + (long)row * NFEAT;

  v8f acc[4] = {};
#pragma unroll
  for (int kt = 0; kt < NFEAT / 32; ++kt) {
    const int k0 = kt * 32;
    __builtin_prefetch(xrow + k0 + 32, 0, 0);          // stream next K block
    float4 f0 = *(const float4*)(xrow + k0 + hi * 16);
    float4 f1 = *(const float4*)(xrow + k0 + hi * 16 + 4);
    float4 f2 = *(const float4*)(xrow + k0 + hi * 16 + 8);
    float4 f3 = *(const float4*)(xrow + k0 + hi * 16 + 12);
    v16h a;
    a[0]  = (_Float16)f0.x; a[1]  = (_Float16)f0.y;
    a[2]  = (_Float16)f0.z; a[3]  = (_Float16)f0.w;
    a[4]  = (_Float16)f1.x; a[5]  = (_Float16)f1.y;
    a[6]  = (_Float16)f1.z; a[7]  = (_Float16)f1.w;
    a[8]  = (_Float16)f2.x; a[9]  = (_Float16)f2.y;
    a[10] = (_Float16)f2.z; a[11] = (_Float16)f2.w;
    a[12] = (_Float16)f3.x; a[13] = (_Float16)f3.y;
    a[14] = (_Float16)f3.z; a[15] = (_Float16)f3.w;

    const int kk = k0 + hi * 16;      // B: lane<16 -> K=k0..k0+15, else +16
#pragma unroll
    for (int nt = 0; nt < 4; ++nt) {
      v16h b = *(const v16h*)&Wt[(nt * 16 + lm) * NFEAT + kk];
      acc[nt] = __builtin_amdgcn_wmma_f32_16x16x32_f16(
          false, a, false, b, (short)0, acc[nt], false, false);
    }
  }
  // C/D layout: VGPR v -> row tile*16 + v + 8*hi, col = n-tile base + lm.
#pragma unroll
  for (int nt = 0; nt < 4; ++nt)
#pragma unroll
    for (int v = 0; v < 8; ++v)
      xw[(long)(tile * 16 + v + hi * 8) * NHID + nt * 16 + lm] = acc[nt][v];
}

// ---------------------------------------------------------------------------
// SpMM1: agg1[dst] += val * xw[src]  over 64 cols. 16 lanes per edge, float4
// gather + 4 f32 atomics (resolve in the 192MB L2; xw is fully L2-resident).
// ---------------------------------------------------------------------------
__global__ __launch_bounds__(256) void k_spmm1(const int* __restrict__ src,
                                               const int* __restrict__ dst,
                                               const float* __restrict__ val,
                                               const float* __restrict__ in,
                                               float* __restrict__ out) {
  long t = (long)blockIdx.x * blockDim.x + threadIdx.x;
  long e = t >> 4;
  if (e >= NEDGE) return;
  int l = (int)(t & 15);
  int s = src[e], d = dst[e];
  float w = val[e];
  float4 v = *(const float4*)(in + (long)s * NHID + l * 4);
  float* o = out + (long)d * NHID + l * 4;
  atomicAdd(o + 0, w * v.x);
  atomicAdd(o + 1, w * v.y);
  atomicAdd(o + 2, w * v.z);
  atomicAdd(o + 3, w * v.w);
}

// ---------------------------------------------------------------------------
// Fused heads GEMM: hw[N,80] = relu(agg1 + b1) @ [W11 | W12]  (64x80 f16 in
// LDS, column-major). relu+bias folded into the A-fragment build so h is
// never materialized. One wave -> 16x80 tile (5 n-tiles, K=64 = 2 WMMA steps).
// ---------------------------------------------------------------------------
__global__ __launch_bounds__(256) void k_heads(const float* __restrict__ agg1,
                                               const float* __restrict__ b1,
                                               const float* __restrict__ W11,
                                               const float* __restrict__ W12,
                                               float* __restrict__ hw) {
  __shared__ __align__(32) _Float16 Wt[NHEAD * NHID];  // Wt[c*64 + r]
  __shared__ float b1s[NHID];
  const int tid = threadIdx.x;
  for (int i = tid; i < NHEAD * NHID; i += 256) {
    int c = i >> 6, r = i & 63;
    float v = (c < NCLS) ? W11[r * NCLS + c] : W12[r * NCLS + (c - NCLS)];
    Wt[c * NHID + r] = (_Float16)v;
  }
  if (tid < NHID) b1s[tid] = b1[tid];
  __syncthreads();

  const int wave = tid >> 5, lane = tid & 31;
  const int tile = blockIdx.x * 8 + wave;
  if (tile >= MTILE) return;

  const int lm  = lane & 15;
  const int hi  = lane >> 4;
  const int klo = hi * 8;
  const int row = tile * 16 + lm;
  const float* hrow = agg1 + (long)row * NHID;

  v8f acc[5] = {};
#pragma unroll
  for (int kt = 0; kt < 2; ++kt) {
    const int k0 = kt * 32;
    v16h a;
#pragma unroll
    for (int j = 0; j < 8; ++j) {
      float t0 = hrow[k0 + hi * 16 + j] + b1s[k0 + hi * 16 + j];
      float t1 = hrow[k0 + hi * 16 + 8 + j] + b1s[k0 + hi * 16 + 8 + j];
      a[j]     = (_Float16)(t0 > 0.f ? t0 : 0.f);
      a[8 + j] = (_Float16)(t1 > 0.f ? t1 : 0.f);
    }
    const int kk = k0 + hi * 16;
#pragma unroll
    for (int nt = 0; nt < 5; ++nt) {
      v16h b = *(const v16h*)&Wt[(nt * 16 + lm) * NHID + kk];
      acc[nt] = __builtin_amdgcn_wmma_f32_16x16x32_f16(
          false, a, false, b, (short)0, acc[nt], false, false);
    }
  }
#pragma unroll
  for (int nt = 0; nt < 5; ++nt)
#pragma unroll
    for (int v = 0; v < 8; ++v)
      hw[(long)(tile * 16 + v + hi * 8) * NHEAD + nt * 16 + lm] = acc[nt][v];
}

// ---------------------------------------------------------------------------
// SpMM2: agg2[dst] += val * hw[src] over 80 cols. 20 col-groups of 4 / edge.
// ---------------------------------------------------------------------------
__global__ __launch_bounds__(256) void k_spmm2(const int* __restrict__ src,
                                               const int* __restrict__ dst,
                                               const float* __restrict__ val,
                                               const float* __restrict__ in,
                                               float* __restrict__ out) {
  long t = (long)blockIdx.x * blockDim.x + threadIdx.x;
  long e = t / 20;
  if (e >= NEDGE) return;
  int g = (int)(t - e * 20);
  int s = src[e], d = dst[e];
  float w = val[e];
  float4 v = *(const float4*)(in + (long)s * NHEAD + g * 4);
  float* o = out + (long)d * NHEAD + g * 4;
  atomicAdd(o + 0, w * v.x);
  atomicAdd(o + 1, w * v.y);
  atomicAdd(o + 2, w * v.z);
  atomicAdd(o + 3, w * v.w);
}

// ---------------------------------------------------------------------------
// z = eps*exp(logstd)+mean, row-wise log_softmax over 40 classes.
// One wave32 per row: lane owns class c=lane (and c=lane+32 for lane<8);
// max/sum reductions via xor shuffles.
// ---------------------------------------------------------------------------
__global__ __launch_bounds__(256) void k_final(const float* __restrict__ agg2,
                                               const float* __restrict__ b11,
                                               const float* __restrict__ b12,
                                               const float* __restrict__ eps,
                                               float* __restrict__ out) {
  const int wave = threadIdx.x >> 5, lane = threadIdx.x & 31;
  const int row = blockIdx.x * 8 + wave;
  if (row >= NNODE) return;
  const float* ar = agg2 + (long)row * NHEAD;
  const float* er = eps + (long)row * NCLS;

  int c1 = lane;
  float z1 = er[c1] * expf(ar[NCLS + c1] + b12[c1]) + (ar[c1] + b11[c1]);
  const bool has2 = lane < (NCLS - 32);
  float z2 = 0.f;
  if (has2) {
    int c2 = lane + 32;
    z2 = er[c2] * expf(ar[NCLS + c2] + b12[c2]) + (ar[c2] + b11[c2]);
  }
  float m = has2 ? fmaxf(z1, z2) : z1;
#pragma unroll
  for (int off = 16; off > 0; off >>= 1) m = fmaxf(m, __shfl_xor(m, off, 32));
  float s = expf(z1 - m) + (has2 ? expf(z2 - m) : 0.f);
#pragma unroll
  for (int off = 16; off > 0; off >>= 1) s += __shfl_xor(s, off, 32);
  float lse = m + logf(s);

  out[(long)row * NCLS + c1] = z1 - lse;
  if (has2) out[(long)row * NCLS + lane + 32] = z2 - lse;
}

// ---------------------------------------------------------------------------
extern "C" void kernel_launch(void* const* d_in, const int* in_sizes, int n_in,
                              void* d_out, int out_size, void* d_ws, size_t ws_size,
                              hipStream_t stream) {
  const float* x    = (const float*)d_in[0];
  const int*   esrc = (const int*)d_in[1];
  const int*   edst = (const int*)d_in[2];
  const float* eval = (const float*)d_in[3];
  const float* eps  = (const float*)d_in[4];
  const float* W1   = (const float*)d_in[5];
  const float* b1   = (const float*)d_in[6];
  const float* W11  = (const float*)d_in[7];
  const float* b11  = (const float*)d_in[8];
  const float* W12  = (const float*)d_in[9];
  const float* b12  = (const float*)d_in[10];
  float* out = (float*)d_out;

  // ws layout (floats): [ xw/hw overlay : 4.0M ][ agg1 : 3.2M ][ agg2 : 4.0M ]
  // xw (N*64) is dead after SpMM1, so hw (N*80) reuses the same region.
  float* ws   = (float*)d_ws;
  float* xw   = ws;                                 // N*NHID  (first 3.2M)
  float* hw   = ws;                                 // N*NHEAD (same region)
  float* agg1 = ws + (size_t)NNODE * NHEAD;         // N*NHID
  float* agg2 = agg1 + (size_t)NNODE * NHID;        // N*NHEAD

  const long nzero = (long)NNODE * (NHID + NHEAD);  // agg1+agg2 contiguous
  k_zero<<<(unsigned)((nzero / 4 + 255) / 256), 256, 0, stream>>>(agg1, nzero);

  k_gemm1<<<(MTILE + 7) / 8, 256, 0, stream>>>(x, W1, xw);

  k_spmm1<<<(unsigned)(((long)NEDGE * 16 + 255) / 256), 256, 0, stream>>>(
      esrc, edst, eval, xw, agg1);

  k_heads<<<(MTILE + 7) / 8, 256, 0, stream>>>(agg1, b1, W11, W12, hw);

  k_spmm2<<<(unsigned)(((long)NEDGE * 20 + 255) / 256), 256, 0, stream>>>(
      esrc, edst, eval, hw, agg2);

  k_final<<<(NNODE + 7) / 8, 256, 0, stream>>>(agg2, b11, b12, eps, out);
}
